// LlamaAttention_6863357739348
// MI455X (gfx1250) — matrix operand order repaired and checked by reading, not compiled, not run on hardware
//
#include <hip/hip_runtime.h>
#include <stdint.h>

typedef __bf16 bf16;
typedef __attribute__((ext_vector_type(16))) bf16 v16bf;
typedef __attribute__((ext_vector_type(8)))  bf16 v8bf;
typedef __attribute__((ext_vector_type(8)))  float v8f;

#define S_  2048
#define E_  4096
#define H_  32
#define KV_ 8
#define D_  128
#define GQ_ (H_ / KV_)   // 4 query heads per kv head

// ---------------- async global->LDS staging (gfx1250 ASYNCcnt path) ---------

#if defined(__has_builtin)
#  if __has_builtin(__builtin_amdgcn_global_load_async_to_lds_b128)
#    define USE_ASYNC_LDS 1
#  endif
#endif
#ifndef USE_ASYNC_LDS
#  define USE_ASYNC_LDS 0
#endif

#if USE_ASYNC_LDS
typedef __attribute__((ext_vector_type(4))) int v4i;
typedef __attribute__((address_space(1))) v4i* gv4i;  // global int4*
typedef __attribute__((address_space(3))) v4i* lv4i;  // LDS int4*
#endif

// copy 16 bf16 (32 bytes) global -> LDS
__device__ __forceinline__ void copy16_g2l(const bf16* __restrict__ g, bf16* l) {
#if USE_ASYNC_LDS
  __builtin_amdgcn_global_load_async_to_lds_b128((gv4i)g, (lv4i)l, 0, 0);
  __builtin_amdgcn_global_load_async_to_lds_b128((gv4i)(g + 8), (lv4i)(l + 8),
                                                 0, 0);
#else
  *(v16bf*)l = *(const v16bf*)g;
#endif
}

__device__ __forceinline__ void async_join() {
#if USE_ASYNC_LDS
#  if defined(__has_builtin) && __has_builtin(__builtin_amdgcn_s_wait_asynccnt)
  __builtin_amdgcn_s_wait_asynccnt(0);
#  else
  asm volatile("s_wait_asynccnt 0x0" ::: "memory");
#  endif
#endif
}

// ---------------- WMMA fragment helpers (CDNA5 16x16x32 bf16 layouts) -------

// A fragment: 16x32 bf16. Lane L (L<16) holds row M=L, K = 0..15;
// lane L+16 holds row M=L, K = 16..31. base points at (row0, k0).
__device__ __forceinline__ v16bf load_a_frag(const bf16* base, int ld) {
  const int lane = threadIdx.x & 31;
  const bf16* p = base + (size_t)(lane & 15) * ld + ((lane >> 4) << 4);
  v8bf lo = *(const v8bf*)(p);        // K = kb .. kb+7
  v8bf hi = *(const v8bf*)(p + 8);    // K = kb+8 .. kb+15
  v16bf a;
#pragma unroll
  for (int i = 0; i < 8; ++i) { a[i] = lo[i]; a[i + 8] = hi[i]; }
  return a;
}

// B fragment: 32x16 bf16, K-contiguous per column. Lane L (L<16): col N=L,
// K=0..15; lane L+16: col N=L-16, K=16..31. base points at (k0, col0) of a
// K-contiguous tile with column stride ld.
__device__ __forceinline__ v16bf load_b_frag(const bf16* base, int ld) {
  const int lane = threadIdx.x & 31;
  return *(const v16bf*)(base + (size_t)(lane & 15) * ld + ((lane >> 4) << 4));
}

__device__ __forceinline__ v8f wmma_bf16(v16bf a, v16bf b, v8f c) {
  return __builtin_amdgcn_wmma_f32_16x16x32_bf16(false, a, false, b,
                                                 (short)0, c, false, false);
}

// ---------------- bf16 tiled GEMM: C[M,N] = A[M,K] * Bt[N,K]^T + bias -------
// Block: 128x128 tile, 256 threads = 8 waves in a 4(M) x 2(N) grid,
// wave tile 32x64 = 2x4 WMMA subtiles, BK = 64, double-buffered LDS.

#define BM 128
#define BN 128
#define BK 64

__global__ __launch_bounds__(256) void gemm_bf16_kernel(
    const bf16* __restrict__ A, const bf16* __restrict__ Bt,
    const float* __restrict__ bias, float* __restrict__ C,
    int M, int N, int K) {
  __shared__ __align__(128) bf16 As[2][BM][BK];   // 2 x 16 KB
  __shared__ __align__(128) bf16 Bs[2][BN][BK];   // 2 x 16 KB
  const int tid  = threadIdx.x;
  const int wave = tid >> 5;
  const int wm   = wave & 3;        // 0..3 -> M offset 32*wm
  const int wn   = wave >> 2;       // 0..1 -> N offset 64*wn
  const int m0   = blockIdx.y * BM;
  const int n0   = blockIdx.x * BN;

  v8f acc[2][4] = {};

  const int trow = tid >> 1;            // 0..127
  const int tkk  = (tid & 1) << 5;      // 0 or 32 (elements)

  auto stage = [&](int buf, int k0) {
    const bf16* ga = A  + (size_t)(m0 + trow) * K + k0 + tkk;
    const bf16* gb = Bt + (size_t)(n0 + trow) * K + k0 + tkk;
    copy16_g2l(ga,      &As[buf][trow][tkk]);
    copy16_g2l(ga + 16, &As[buf][trow][tkk + 16]);
    copy16_g2l(gb,      &Bs[buf][trow][tkk]);
    copy16_g2l(gb + 16, &Bs[buf][trow][tkk + 16]);
  };

  stage(0, 0);
  async_join();
  __syncthreads();

  int buf = 0;
  for (int k0 = 0; k0 < K; k0 += BK) {
    if (k0 + BK < K) stage(buf ^ 1, k0 + BK);       // overlap with compute
    if (k0 + 2 * BK < K) {
      __builtin_prefetch(A  + (size_t)(m0 + trow) * K + k0 + 2 * BK + tkk, 0, 1);
      __builtin_prefetch(Bt + (size_t)(n0 + trow) * K + k0 + 2 * BK + tkk, 0, 1);
    }
#pragma unroll
    for (int kk = 0; kk < BK; kk += 32) {
      v16bf a0 = load_a_frag(&As[buf][wm * 32][kk],      BK);
      v16bf a1 = load_a_frag(&As[buf][wm * 32 + 16][kk], BK);
      v16bf b4[4];
#pragma unroll
      for (int j = 0; j < 4; ++j)
        b4[j] = load_b_frag(&Bs[buf][wn * 64 + j * 16][kk], BK);
#pragma unroll
      for (int j = 0; j < 4; ++j) {
        acc[0][j] = wmma_bf16(a0, b4[j], acc[0][j]);
        acc[1][j] = wmma_bf16(a1, b4[j], acc[1][j]);
      }
    }
    async_join();
    __syncthreads();
    buf ^= 1;
  }

  const int lane = tid & 31;
  const int cl   = lane & 15;
  const int hi   = lane >> 4;
#pragma unroll
  for (int sm = 0; sm < 2; ++sm) {
#pragma unroll
    for (int j = 0; j < 4; ++j) {
      const int col  = n0 + wn * 64 + j * 16 + cl;
      const int row0 = m0 + wm * 32 + sm * 16 + hi * 8;
      const float bv = bias[col];
#pragma unroll
      for (int r = 0; r < 8; ++r)
        C[(size_t)(row0 + r) * N + col] = acc[sm][j][r] + bv;
    }
  }
}

// ---------------- elementwise converters ------------------------------------

__global__ void cvt_kernel(const float* __restrict__ x, bf16* __restrict__ y,
                           long n) {
  long i = (long)blockIdx.x * blockDim.x + threadIdx.x;
  if (i < n) y[i] = (bf16)x[i];
}

// w[K][N] fp32 -> wt[N][K] bf16 (so GEMM B loads are K-contiguous)
__global__ void tcvt_kernel(const float* __restrict__ w, bf16* __restrict__ wt,
                            int K, int N) {
  long i = (long)blockIdx.x * blockDim.x + threadIdx.x;
  if (i < (long)K * N) {
    int k = (int)(i / N), n = (int)(i % N);
    wt[(size_t)n * K + k] = (bf16)w[i];
  }
}

// ---------------- RoPE (fp32 in, bf16 head-major out) -----------------------
// inp: [S][nh*D] fp32, outp: [nh][S][D] bf16. position_ids == arange(S).
__global__ void rope_kernel(const float* __restrict__ inp,
                            bf16* __restrict__ outp, int nh) {
  const int s = blockIdx.x, hh = blockIdx.y, d = threadIdx.x;
  const float* row = inp + ((size_t)s * nh + hh) * D_;
  const float xv = row[d];
  const float xr = (d < 64) ? -row[d + 64] : row[d - 64];
  const float ang =
      (float)s * __powf(10000.0f, -(float)(2 * (d & 63)) * (1.0f / 128.0f));
  const float c = __cosf(ang), sn = __sinf(ang);
  outp[((size_t)hh * S_ + s) * D_ + d] = (bf16)(xv * c + xr * sn);
}

// v fp32 [S][KV*D] -> v^T bf16 [KV][D][S] (key-contiguous for PV B-fragments)
__global__ void vconv_kernel(const float* __restrict__ vf,
                             bf16* __restrict__ vtb) {
  const int s = blockIdx.x, kv = blockIdx.y, d = threadIdx.x;
  vtb[((size_t)kv * D_ + d) * S_ + s] = (bf16)vf[((size_t)s * KV_ + kv) * D_ + d];
}

// ---------------- flash attention -------------------------------------------
// Block: 1 head x 128 query rows (16 per wave). Iterate 64-key blocks with
// online softmax; all matmuls via WMMA bf16.

#define BR 128
#define BC 64

__global__ __launch_bounds__(256) void attn_kernel(
    const bf16* __restrict__ qb,    // [H][S][D]
    const bf16* __restrict__ kbuf,  // [KV][S][D]
    const bf16* __restrict__ vtb,   // [KV][D][S]
    bf16* __restrict__ ctxb) {      // [S][H*D]
  __shared__ __align__(128) bf16 Ks[BC][D_];     // 16 KB  [key][d]
  __shared__ __align__(128) bf16 Vts[D_][BC];    // 16 KB  [d][key]
  __shared__ __align__(128) bf16 Ps[8][16][BC];  // 16 KB  per-wave P staging
  const int h    = blockIdx.y;
  const int kv   = h / GQ_;
  const int q0   = blockIdx.x * BR;
  const int tid  = threadIdx.x;
  const int wave = tid >> 5;
  const int lane = tid & 31;
  const int cl   = lane & 15;
  const int hi   = lane >> 4;
  const int r0   = q0 + wave * 16;

  // Preload this wave's Q fragments: 16 rows x 128 dims = 4 K-chunks of 32.
  const bf16* qbase = qb + ((size_t)h * S_ + r0) * D_;
  v16bf qa[4];
#pragma unroll
  for (int ks = 0; ks < 4; ++ks) qa[ks] = load_a_frag(qbase + ks * 32, D_);

  float m_[8], l_[8];
  v8f o[8] = {};
#pragma unroll
  for (int r = 0; r < 8; ++r) { m_[r] = -1e30f; l_[r] = 0.f; }

  const float scale = 0.08838834764831845f;  // 1/sqrt(128)

  for (int j0 = 0; j0 < q0 + BR; j0 += BC) {  // causal: keys <= last row
    {  // stage K block [BC][D_]
      const int row = tid >> 2;
      const int kk  = (tid & 3) << 5;
      const bf16* src = kbuf + ((size_t)kv * S_ + j0 + row) * D_ + kk;
      copy16_g2l(src,      &Ks[row][kk]);
      copy16_g2l(src + 16, &Ks[row][kk + 16]);
    }
    {  // stage V^T block [D_][BC]
      const int d  = tid >> 1;
      const int kk = (tid & 1) << 5;
      const bf16* src = vtb + ((size_t)kv * D_ + d) * S_ + j0 + kk;
      copy16_g2l(src,      &Vts[d][kk]);
      copy16_g2l(src + 16, &Vts[d][kk + 16]);
    }
    async_join();
    __syncthreads();

    // scores S = Q K^T : 16 rows x 64 keys per wave, 4 K-chunks over d
    v8f sc[4] = {};
#pragma unroll
    for (int ks = 0; ks < 4; ++ks) {
      v16bf b4[4];
#pragma unroll
      for (int j = 0; j < 4; ++j)
        b4[j] = load_b_frag(&Ks[j * 16][ks * 32], D_);
#pragma unroll
      for (int j = 0; j < 4; ++j) sc[j] = wmma_bf16(qa[ks], b4[j], sc[j]);
    }

    // online softmax: a row spans the 16 lanes of one half-wave
#pragma unroll
    for (int r = 0; r < 8; ++r) {
      const int rowg = r0 + r + hi * 8;
      float mx = -3.0e38f;
#pragma unroll
      for (int j = 0; j < 4; ++j) {
        const int colg = j0 + j * 16 + cl;
        float v = sc[j][r] * scale;
        if (colg > rowg) v = -3.0e38f;  // causal mask
        sc[j][r] = v;
        mx = fmaxf(mx, v);
      }
#pragma unroll
      for (int off = 8; off > 0; off >>= 1)
        mx = fmaxf(mx, __shfl_xor(mx, off, 16));
      const float mn   = fmaxf(m_[r], mx);
      const float corr = __expf(m_[r] - mn);
      float rsum = 0.f;
#pragma unroll
      for (int j = 0; j < 4; ++j) {
        const float pv = __expf(sc[j][r] - mn);
        rsum += pv;
        Ps[wave][r + hi * 8][j * 16 + cl] = (bf16)pv;  // C-layout -> LDS
      }
#pragma unroll
      for (int off = 8; off > 0; off >>= 1)
        rsum += __shfl_xor(rsum, off, 16);
      l_[r] = l_[r] * corr + rsum;
      m_[r] = mn;
#pragma unroll
      for (int n = 0; n < 8; ++n) o[n][r] *= corr;
    }

    // O += P @ V : reload P from LDS as A-fragments (K = 64 keys = 2 chunks)
#pragma unroll
    for (int kc = 0; kc < 2; ++kc) {
      v16bf pa = load_a_frag(&Ps[wave][0][kc * 32], BC);
#pragma unroll
      for (int half = 0; half < 2; ++half) {
        v16bf b4[4];
#pragma unroll
        for (int j = 0; j < 4; ++j)
          b4[j] = load_b_frag(&Vts[(half * 4 + j) * 16][kc * 32], BC);
#pragma unroll
        for (int j = 0; j < 4; ++j)
          o[half * 4 + j] = wmma_bf16(pa, b4[j], o[half * 4 + j]);
      }
    }
    __syncthreads();
  }

  // normalize and write context bf16 [S][H*D]
#pragma unroll
  for (int r = 0; r < 8; ++r) {
    const float inv  = 1.f / l_[r];
    const int   rowg = r0 + r + hi * 8;
#pragma unroll
    for (int n = 0; n < 8; ++n)
      ctxb[(size_t)rowg * (H_ * D_) + h * D_ + n * 16 + cl] =
          (bf16)(o[n][r] * inv);
  }
}

// ---------------- launch -----------------------------------------------------

extern "C" void kernel_launch(void* const* d_in, const int* in_sizes, int n_in,
                              void* d_out, int out_size, void* d_ws,
                              size_t ws_size, hipStream_t stream) {
  (void)in_sizes; (void)n_in; (void)out_size; (void)ws_size;
  const float* x  = (const float*)d_in[0];
  // d_in[1] = mask (causal applied analytically), d_in[2] = position_ids (= arange)
  const float* wq = (const float*)d_in[3];
  const float* bq = (const float*)d_in[4];
  const float* wk = (const float*)d_in[5];
  const float* bk = (const float*)d_in[6];
  const float* wv = (const float*)d_in[7];
  const float* bv = (const float*)d_in[8];
  const float* wo = (const float*)d_in[9];
  const float* bo = (const float*)d_in[10];
  float* out = (float*)d_out;

  uint8_t* p = (uint8_t*)d_ws;
  auto take = [&](size_t bytes) {
    uint8_t* q = p;
    p += (bytes + 255) & ~(size_t)255;
    return q;
  };
  bf16*  xb   = (bf16*)take((size_t)S_ * E_ * 2);
  bf16*  wqT  = (bf16*)take((size_t)E_ * H_ * D_ * 2);
  bf16*  wkT  = (bf16*)take((size_t)E_ * KV_ * D_ * 2);
  bf16*  wvT  = (bf16*)take((size_t)E_ * KV_ * D_ * 2);
  bf16*  woT  = (bf16*)take((size_t)H_ * D_ * E_ * 2);
  float* qf   = (float*)take((size_t)S_ * H_ * D_ * 4);
  float* kf   = (float*)take((size_t)S_ * KV_ * D_ * 4);
  float* vf   = (float*)take((size_t)S_ * KV_ * D_ * 4);
  bf16*  qbb  = (bf16*)take((size_t)H_ * S_ * D_ * 2);
  bf16*  kbb  = (bf16*)take((size_t)KV_ * S_ * D_ * 2);
  bf16*  vtb  = (bf16*)take((size_t)KV_ * D_ * S_ * 2);
  bf16*  ctxb = (bf16*)take((size_t)S_ * H_ * D_ * 2);

  const long nx = (long)S_ * E_;
  cvt_kernel<<<(unsigned)((nx + 255) / 256), 256, 0, stream>>>(x, xb, nx);
  tcvt_kernel<<<(unsigned)(((long)E_ * H_ * D_ + 255) / 256), 256, 0, stream>>>(
      wq, wqT, E_, H_ * D_);
  tcvt_kernel<<<(unsigned)(((long)E_ * KV_ * D_ + 255) / 256), 256, 0, stream>>>(
      wk, wkT, E_, KV_ * D_);
  tcvt_kernel<<<(unsigned)(((long)E_ * KV_ * D_ + 255) / 256), 256, 0, stream>>>(
      wv, wvT, E_, KV_ * D_);
  tcvt_kernel<<<(unsigned)(((long)H_ * D_ * E_ + 255) / 256), 256, 0, stream>>>(
      wo, woT, H_ * D_, E_);

  gemm_bf16_kernel<<<dim3((H_ * D_) / BN, S_ / BM), 256, 0, stream>>>(
      xb, wqT, bq, qf, S_, H_ * D_, E_);
  gemm_bf16_kernel<<<dim3((KV_ * D_) / BN, S_ / BM), 256, 0, stream>>>(
      xb, wkT, bk, kf, S_, KV_ * D_, E_);
  gemm_bf16_kernel<<<dim3((KV_ * D_) / BN, S_ / BM), 256, 0, stream>>>(
      xb, wvT, bv, vf, S_, KV_ * D_, E_);

  rope_kernel<<<dim3(S_, H_), D_, 0, stream>>>(qf, qbb, H_);
  rope_kernel<<<dim3(S_, KV_), D_, 0, stream>>>(kf, kbb, KV_);
  vconv_kernel<<<dim3(S_, KV_), D_, 0, stream>>>(vf, vtb);

  attn_kernel<<<dim3(S_ / BR, H_), 256, 0, stream>>>(qbb, kbb, vtb, ctxb);

  gemm_bf16_kernel<<<dim3(E_ / BN, S_ / BM), 256, 0, stream>>>(
      ctxb, woT, bo, out, S_, E_, H_ * D_);
}
